// SRUCell_77936476553434
// MI455X (gfx1250) — matrix operand order repaired and checked by reading, not compiled, not run on hardware
//
#include <hip/hip_runtime.h>
#include <hip/hip_bf16.h>

// SRU forward for MI455X (gfx1250).
// Phase 0  (once/launch): weight fp32 -> bf16 hi/lo planes, packed per-(nblock,kstage)
//                         in the exact LDS tile image (K-contiguous columns).
// Phase 0b (per chunk):   input chunk fp32 -> bf16 hi/lo planes, packed per-(mblock,kstage)
//                         K-contiguous rows (converted ONCE, not once per N-block).
// Phase 1  (per chunk):   U = A @ W via bf16x3-split V_WMMA_F32_16X16X32_BF16; both
//                         operands streamed into double-buffered LDS with
//                         global_load_async_to_lds_b128 (ASYNCcnt), 1 barrier/stage.
// Phase 2  (per chunk):   sequential gating scan (bandwidth bound), software-pipelined.

typedef __attribute__((ext_vector_type(4)))  float  v4f;
typedef __attribute__((ext_vector_type(8)))  float  v8f;
typedef __attribute__((ext_vector_type(4)))  __bf16 v4bf;
typedef __attribute__((ext_vector_type(8)))  __bf16 v8bf;
typedef __attribute__((ext_vector_type(16))) __bf16 v16bf;

#define SEQ_LEN 2048
#define BATCH   16
#define HID     1024
#define GK      1024
#define GN      3072
#define BM      128
#define BN      128
#define BK      32        // K per LDS stage = one bf16 WMMA K
#define NSTAGE  (GK / BK) // 32
#define LDSTR   40        // padded LDS stride (bf16): 80B, 16B-aligned, conflict-free

__device__ __forceinline__ void cvt_split4(v4f v, v4bf& hi, v4bf& lo) {
#pragma unroll
    for (int j = 0; j < 4; ++j) {
        float f  = v[j];
        __bf16 h = (__bf16)f;
        hi[j] = h;
        lo[j] = (__bf16)(f - (float)h);
    }
}

__device__ __forceinline__ v16bf join16(v8bf a, v8bf b) {
    return __builtin_shufflevector(a, b, 0, 1, 2, 3, 4, 5, 6, 7,
                                         8, 9, 10, 11, 12, 13, 14, 15);
}

// Async 16B/lane copy global -> LDS (gfx1250), GVS addressing: saddr base + vgpr offset.
__device__ __forceinline__ void async_b128(unsigned lds_byte_off, const void* sbase, int voff) {
    asm volatile("global_load_async_to_lds_b128 %0, %1, %2"
                 :: "v"(lds_byte_off), "v"(voff), "s"(sbase)
                 : "memory");
}
__device__ __forceinline__ void wait_async0() {
    asm volatile("s_wait_asynccnt 0x0" ::: "memory");
}
__device__ __forceinline__ unsigned lds_off32(const void* p) {
    return (unsigned)(size_t)p;   // LDS aperture: low 32 bits = LDS byte offset
}

// Fragment-order permutation of K within a 32-chunk: [K0-7 | K16-23 | K8-15 | K24-31]
__device__ __forceinline__ int frag_pos(int kk) {
    int g  = kk >> 3;
    int pg = (g == 1) ? 2 : ((g == 2) ? 1 : g);
    return (pg << 3) + (kk & 7);
}

// -------------------- Phase 0: split/pack weight --------------------
// off = ((bn*NSTAGE + s)*BN + n)*BK + k   (column-of-tile major, K contiguous per column)
__global__ __launch_bounds__(256)
void sru_prep_w(const float* __restrict__ W, __bf16* __restrict__ Whi, __bf16* __restrict__ Wlo)
{
    const int tid = blockIdx.x * 256 + threadIdx.x;   // 0 .. 1024*768-1
    const int kg  = tid / (GN / 4);
    const int c4  = (tid % (GN / 4)) * 4;
    v4f w = *(const v4f*)(W + (size_t)kg * GN + c4);
    v4bf hi, lo;
    cvt_split4(w, hi, lo);
    const int s = kg >> 5, k = kg & 31;
#pragma unroll
    for (int j = 0; j < 4; ++j) {
        const int col = c4 + j;
        const int bn  = col >> 7, n = col & 127;
        const size_t off = (((size_t)bn * NSTAGE + s) * BN + n) * BK + k;
        Whi[off] = hi[j];
        Wlo[off] = lo[j];
    }
}

// -------------------- Phase 0b: split/pack input chunk --------------------
// off = ((bm*NSTAGE + s)*BM + m)*BK + k   (row-of-tile major, K contiguous per row)
__global__ __launch_bounds__(256)
void sru_prep_a(const float* __restrict__ A, __bf16* __restrict__ Ahi, __bf16* __restrict__ Alo)
{
    const int tid = blockIdx.x * 256 + threadIdx.x;   // R*256 threads (R = rows)
    const int r   = tid >> 8;                         // row
    const int kq  = (tid & 255) << 2;                 // K of first of 4
    v4f a = *(const v4f*)(A + (size_t)r * GK + kq);
    v4bf hi, lo;
    cvt_split4(a, hi, lo);
    const int bm = r >> 7, m = r & 127;
    const int s  = kq >> 5;
    const int pos = kq & 31;                          // 4 consecutive within an 8-group
    const size_t off = (((size_t)bm * NSTAGE + s) * BM + m) * BK + pos;
    *(v4bf*)(Ahi + off) = hi;
    *(v4bf*)(Alo + off) = lo;
}

// -------------------- Phase 1: GEMM U = A * W (bf16x3 WMMA, fully async-staged) ----------
__global__ __launch_bounds__(256)
void sru_gemm_bf16x3(const __bf16* __restrict__ Ahi, const __bf16* __restrict__ Alo,
                     const __bf16* __restrict__ Whi, const __bf16* __restrict__ Wlo,
                     float* __restrict__ U)
{
    __shared__ __align__(16) __bf16 lds_ahi[2][BM * LDSTR];
    __shared__ __align__(16) __bf16 lds_alo[2][BM * LDSTR];
    __shared__ __align__(16) __bf16 lds_bhi[2][BN * LDSTR];
    __shared__ __align__(16) __bf16 lds_blo[2][BN * LDSTR];

    const int nBlocksN = GN / BN;   // 24
    const int bm = blockIdx.x / nBlocksN;
    const int bn = blockIdx.x % nBlocksN;

    const int t    = threadIdx.x;
    const int lane = t & 31;
    const int wave = t >> 5;
    const int l15  = lane & 15;
    const int lhi  = lane >> 4;

    v8f acc[8] = {};

    // Per-thread async coordinates: 8KB tile/plane = 2 rounds of 16B per thread.
    const int n0  = t >> 2;
    const int wb  = (t & 3) << 4;
    const int vo0 = (n0 << 6) | wb;                 // packed global: 64B per row/col
    const int vo1 = ((n0 + 64) << 6) | wb;
    const unsigned po0 = (unsigned)(n0 * 80 + wb);  // padded LDS: 80B per row/col
    const unsigned po1 = (unsigned)((n0 + 64) * 80 + wb);

    const unsigned ahOff[2] = { lds_off32(&lds_ahi[0][0]), lds_off32(&lds_ahi[1][0]) };
    const unsigned alOff[2] = { lds_off32(&lds_alo[0][0]), lds_off32(&lds_alo[1][0]) };
    const unsigned bhOff[2] = { lds_off32(&lds_bhi[0][0]), lds_off32(&lds_bhi[1][0]) };
    const unsigned blOff[2] = { lds_off32(&lds_blo[0][0]), lds_off32(&lds_blo[1][0]) };

    const __bf16* aBaseH = Ahi + ((size_t)bm * NSTAGE) * BM * BK;
    const __bf16* aBaseL = Alo + ((size_t)bm * NSTAGE) * BM * BK;
    const __bf16* bBaseH = Whi + ((size_t)bn * NSTAGE) * BN * BK;
    const __bf16* bBaseL = Wlo + ((size_t)bn * NSTAGE) * BN * BK;

    // Issue stage 0 into buffer 0 (A and B, hi and lo).
#define ISSUE_STAGE(buf, s_)                                                        \
    do {                                                                            \
        const char* ah = (const char*)(aBaseH + (size_t)(s_) * BM * BK);            \
        const char* al = (const char*)(aBaseL + (size_t)(s_) * BM * BK);            \
        const char* bh = (const char*)(bBaseH + (size_t)(s_) * BN * BK);            \
        const char* bl = (const char*)(bBaseL + (size_t)(s_) * BN * BK);            \
        async_b128(ahOff[buf] + po0, ah, vo0); async_b128(ahOff[buf] + po1, ah, vo1);\
        async_b128(alOff[buf] + po0, al, vo0); async_b128(alOff[buf] + po1, al, vo1);\
        async_b128(bhOff[buf] + po0, bh, vo0); async_b128(bhOff[buf] + po1, bh, vo1);\
        async_b128(blOff[buf] + po0, bl, vo0); async_b128(blOff[buf] + po1, bl, vo1);\
    } while (0)

    ISSUE_STAGE(0, 0);

    for (int s = 0; s < NSTAGE; ++s) {
        const int cur = s & 1, nxt = cur ^ 1;

        wait_async0();          // this wave's stage-s tiles have landed
        __syncthreads();        // all waves' tiles visible; prior reads of `nxt` done

        if (s + 1 < NSTAGE) ISSUE_STAGE(nxt, s + 1);

        // 8 N-tiles x 3 split WMMAs over this K=32 chunk.
        const int afo = (wave * 16 + l15) * LDSTR + (lhi << 4);
        const __bf16* ah = lds_ahi[cur];
        const __bf16* al = lds_alo[cur];
        v16bf ahi = join16(*(const v8bf*)(&ah[afo]), *(const v8bf*)(&ah[afo + 8]));
        v16bf alo = join16(*(const v8bf*)(&al[afo]), *(const v8bf*)(&al[afo + 8]));
        const __bf16* bh = lds_bhi[cur];
        const __bf16* bl = lds_blo[cur];
#pragma unroll
        for (int tn = 0; tn < 8; ++tn) {
            const int bfo = (tn * 16 + l15) * LDSTR + (lhi << 4);
            v16bf bhi = join16(*(const v8bf*)(&bh[bfo]), *(const v8bf*)(&bh[bfo + 8]));
            v16bf blo = join16(*(const v8bf*)(&bl[bfo]), *(const v8bf*)(&bl[bfo + 8]));
            acc[tn] = __builtin_amdgcn_wmma_f32_16x16x32_bf16(
                false, ahi, false, bhi, (short)0, acc[tn], false, false);
            acc[tn] = __builtin_amdgcn_wmma_f32_16x16x32_bf16(
                false, ahi, false, blo, (short)0, acc[tn], false, false);
            acc[tn] = __builtin_amdgcn_wmma_f32_16x16x32_bf16(
                false, alo, false, bhi, (short)0, acc[tn], false, false);
        }
    }
#undef ISSUE_STAGE

    const int rowBase = bm * BM, colBase = bn * BN;
#pragma unroll
    for (int tn = 0; tn < 8; ++tn) {
        const int col = colBase + tn * 16 + l15;
#pragma unroll
        for (int r = 0; r < 8; ++r) {
            const int row = rowBase + wave * 16 + r + (lhi << 3);
            U[(size_t)row * GN + col] = acc[tn][r];
        }
    }
}

// -------------------- Phase 2: scan --------------------
__global__ __launch_bounds__(256)
void sru_scan(const float* __restrict__ U, const float* __restrict__ X,
              const float* __restrict__ bias,
              float* __restrict__ Hout, float* __restrict__ Cout,
              float* __restrict__ carry, int t0, int nT)
{
    const int tid = blockIdx.x * 256 + threadIdx.x;   // 0..16383
    const int b   = tid >> 10;
    const int h   = tid & (HID - 1);

    const float b0 = bias[h];
    const float b1 = bias[HID + h];

    float c = (t0 == 0) ? 0.0f : carry[tid];

    const size_t ustep = (size_t)BATCH * GN;
    const size_t xstep = (size_t)BATCH * HID;
    size_t ub = (size_t)b * GN + 3 * h;
    size_t xo = ((size_t)t0 * BATCH + b) * HID + h;

    float u0 = U[ub], u1 = U[ub + 1], u2 = U[ub + 2], xt = X[xo];

    for (int tt = 0; tt < nT; ++tt) {
        const float cu0 = u0, cu1 = u1, cu2 = u2, cx = xt;
        const size_t o = xo;
        if (tt + 1 < nT) {
            ub += ustep; xo += xstep;
            u0 = U[ub]; u1 = U[ub + 1]; u2 = U[ub + 2]; xt = X[xo];
        }
        const float f = 1.0f / (1.0f + __expf(-(cu0 + b0 * c + b0)));
        const float r = 1.0f / (1.0f + __expf(-(cu1 + b1 * c + b1)));
        c = f * c + (1.0f - f) * cu2;
        const float ht = r * c + (1.0f - r) * cx;
        Hout[o] = ht;
        Cout[o] = c;
    }
    carry[tid] = c;
}

// -------------------- Host launch --------------------
extern "C" void kernel_launch(void* const* d_in, const int* in_sizes, int n_in,
                              void* d_out, int out_size, void* d_ws, size_t ws_size,
                              hipStream_t stream)
{
    (void)in_sizes; (void)n_in; (void)out_size;

    const float* inp  = (const float*)d_in[0];
    const float* w    = (const float*)d_in[1];
    const float* bias = (const float*)d_in[2];
    // d_in[3] = v : intentionally unused (reference aliases v0,v1 to bias)

    float* H = (float*)d_out;
    float* C = H + (size_t)SEQ_LEN * BATCH * HID;

    // Workspace layout: carry | Whi | Wlo | Ahi | Alo | U(chunk)
    char* ws = (char*)d_ws;
    float*  carry = (float*)ws;
    const size_t carryBytes = (size_t)BATCH * HID * sizeof(float);          // 64 KB
    __bf16* Whi = (__bf16*)(ws + carryBytes);
    const size_t wBytes = (size_t)GK * GN * sizeof(__bf16);                 // 6 MB
    __bf16* Wlo = (__bf16*)(ws + carryBytes + wBytes);

    // Per-timestep workspace: Ahi+Alo (16*1024*2B each) + U (16*3072*4B)
    const size_t aPerT = (size_t)BATCH * GK * sizeof(__bf16);               // 32 KB
    const size_t uPerT = (size_t)BATCH * GN * sizeof(float);                // 192 KB
    const size_t perT  = 2 * aPerT + uPerT;
    const size_t reserve = carryBytes + 2 * wBytes;
    size_t avail = (ws_size > reserve) ? (ws_size - reserve) : 0;
    long maxChunk = (long)(avail / perT);
    int chunk = 8;
    while ((long)chunk * 2 <= maxChunk && chunk < SEQ_LEN) chunk *= 2;

    __bf16* Ahi = (__bf16*)(ws + reserve);
    __bf16* Alo = Ahi + (size_t)chunk * BATCH * GK;
    float*  U   = (float*)(Alo + (size_t)chunk * BATCH * GK);

    // Phase 0: pack weight planes (once per launch).
    sru_prep_w<<<dim3((GK * GN / 4) / 256), 256, 0, stream>>>(w, Whi, Wlo);

    const int nBlocksN = GN / BN;  // 24
    for (int t0 = 0; t0 < SEQ_LEN; t0 += chunk) {
        const int nT = (SEQ_LEN - t0 < chunk) ? (SEQ_LEN - t0) : chunk;
        const int rows = nT * BATCH;
        sru_prep_a<<<dim3(rows), 256, 0, stream>>>(
            inp + (size_t)t0 * BATCH * HID, Ahi, Alo);
        sru_gemm_bf16x3<<<dim3((rows / BM) * nBlocksN), 256, 0, stream>>>(
            Ahi, Alo, Whi, Wlo, U);
        sru_scan<<<dim3((BATCH * HID) / 256), 256, 0, stream>>>(
            U, inp, bias, H, C, carry, t0, nT);
    }
}
